// EfficientSelfAttention_57621281243705
// MI455X (gfx1250) — matrix operand order repaired and checked by reading, not compiled, not run on hardware
//
#include <hip/hip_runtime.h>
#include <hip/hip_bf16.h>

typedef __attribute__((ext_vector_type(16))) _Float16 v16h;
typedef __attribute__((ext_vector_type(8)))  _Float16 v8h;
typedef __attribute__((ext_vector_type(4)))  _Float16 v4h;
typedef __attribute__((ext_vector_type(8)))  float    v8f;

#define WMMA_F32_F16(A, Bm, Cm) \
  __builtin_amdgcn_wmma_f32_16x16x32_f16(false, (A), false, (Bm), (short)0, (Cm), false, false)

// Problem constants
#define BATCH 8
#define CDIM  256
#define NTOK  4096   // 64*64
#define HEADS 8
#define DHEAD 32
#define MTOK  256    // 16*16 after 4x4 stride-4 reduction
#define LDSC  264    // padded c-stride for full LDS B-panels (528B = 33x16B)
#define KVP   136    // padded half-panel stride for kv (272B = 17x16B)
#define SIMP  260    // padded sim row (f32): 1040B, 4-bank spread
#define PSP   264    // padded p row (f16): 528B, 4-bank spread

// Build a v16h WMMA operand from two contiguous, 16B-aligned 8-half chunks.
static __device__ __forceinline__ v16h ld16(const _Float16* lo, const _Float16* hi) {
  v8h a = *(const v8h*)lo;
  v8h b = *(const v8h*)hi;
  return __builtin_shufflevector(a, b, 0, 1, 2, 3, 4, 5, 6, 7,
                                       8, 9, 10, 11, 12, 13, 14, 15);
}

// CDNA5 async global->LDS copy (16B per lane), tracked by ASYNCcnt.
// VDST holds the LDS byte address (low 32 bits of a generic LDS pointer,
// per ISA 10.2: generic shared addr = {aperture_hi, lds_offset}).
static __device__ __forceinline__ void async_b128(void* lds, const void* g) {
  unsigned loff = (unsigned)(size_t)lds;
  unsigned long long ga = (unsigned long long)(size_t)g;
  asm volatile("global_load_async_to_lds_b128 %0, %1, off"
               :: "v"(loff), "v"(ga) : "memory");
}
static __device__ __forceinline__ void wait_async0() {
  asm volatile("s_wait_asynccnt 0" ::: "memory");
}
// Async loads complete in order: waiting for <=8 outstanding guarantees the
// previous 8-op panel is done while the just-issued 8-op panel is in flight.
static __device__ __forceinline__ void wait_async_le8() {
  asm volatile("s_wait_asynccnt 8" ::: "memory");
}

// ---------------------------------------------------------------------------
// Transpose+convert x: [C][N] f32 -> xT [N][C] f16 (per batch). 32x32 tiles.
// ---------------------------------------------------------------------------
__global__ __launch_bounds__(256) void transpose_cvt(const float* __restrict__ src,
                                                     _Float16* __restrict__ dstT) {
  const int b  = blockIdx.z;
  const int c0 = blockIdx.y * 32;
  const int n0 = blockIdx.x * 32;
  __shared__ float tile[32][33];
  const int tx = threadIdx.x & 31;
  const int ty = threadIdx.x >> 5;   // 0..7
#pragma unroll
  for (int j = 0; j < 32; j += 8)
    tile[ty + j][tx] = src[((size_t)b * CDIM + c0 + ty + j) * NTOK + n0 + tx];
  __syncthreads();
#pragma unroll
  for (int j = 0; j < 32; j += 8)
    dstT[((size_t)b * NTOK + n0 + ty + j) * CDIM + c0 + tx] = (_Float16)tile[tx][ty + j];
}

// ---------------------------------------------------------------------------
// fp32 -> fp16 conversion (grid-stride) for w_q / w_out
// ---------------------------------------------------------------------------
__global__ __launch_bounds__(256) void cvt_f32_f16(const float* __restrict__ s,
                                                   _Float16* __restrict__ d, int n) {
  int i = blockIdx.x * blockDim.x + threadIdx.x;
  int stride = gridDim.x * blockDim.x;
  for (; i < n; i += stride) d[i] = (_Float16)s[i];
}

// ---------------------------------------------------------------------------
// Permute+convert w_kv [O][c][i][j] f32 -> wkvP [O][(i*4+j)*256 + c] f16,
// so each conv tap's K-slice is contiguous in c (matches xT rows).
// ---------------------------------------------------------------------------
__global__ __launch_bounds__(256) void permute_wkv(const float* __restrict__ s,
                                                   _Float16* __restrict__ d) {
  int i = blockIdx.x * 256 + threadIdx.x;   // 512*4096 total
  int o  = i >> 12;
  int r  = i & 4095;
  int ij = r >> 8, c = r & 255;
  d[i] = (_Float16)s[(size_t)o * 4096 + c * 16 + ij];
}

// ---------------------------------------------------------------------------
// Q projection: q[o,n] = sum_c w_q[o,c] * xT[n,c]; store split-head, scaled.
// WG = 128 (4 waves). Output tile 64(o) x 64(n). Full 64x256 B-panel staged
// once via async-to-LDS; K-loop runs barrier-free out of LDS.
// ---------------------------------------------------------------------------
__global__ __launch_bounds__(128) void qproj_kernel(const _Float16* __restrict__ xT,
                                                    const _Float16* __restrict__ wqh,
                                                    _Float16* __restrict__ qh) {
  const int b  = blockIdx.z;
  const int o0 = blockIdx.y * 64;
  const int n0 = blockIdx.x * 64;
  const int tid = threadIdx.x;
  const int wave = tid >> 5, lane = tid & 31;

  __shared__ __align__(16) _Float16 Xs[64][LDSC];   // [n][c], full K panel

  // async-stage the whole B panel: 2048 x 16B chunks, 16 per thread
  for (int ch = tid; ch < 2048; ch += 128) {
    int row = ch >> 5;
    int cc  = (ch & 31) * 8;
    async_b128(&Xs[row][cc], &xT[((size_t)b * NTOK + n0 + row) * CDIM + cc]);
  }
  wait_async0();
  __syncthreads();

  v8f acc[4] = {};
  const int arow = o0 + wave * 16 + (lane & 15);
  const int koff = (lane < 16) ? 0 : 8;    // unused; A operand also uses kb
  const int kb   = (lane < 16) ? 0 : 16;   // B-operand K base per lane half

  for (int kc = 0; kc < CDIM; kc += 32) {
    const _Float16* wp = wqh + (size_t)arow * CDIM + kc;
    v16h a = ld16(wp + kb, wp + kb + 8);
#pragma unroll
    for (int nt = 0; nt < 4; ++nt) {
      const _Float16* bp = &Xs[nt * 16 + (lane & 15)][kc + kb];
      v16h bm = ld16(bp, bp + 8);
      acc[nt] = WMMA_F32_F16(a, bm, acc[nt]);
    }
  }

  const float scale = 0.17677669529663687f; // 1/sqrt(32), folded into q
#pragma unroll
  for (int nt = 0; nt < 4; ++nt) {
#pragma unroll
    for (int r = 0; r < 8; ++r) {
      int o = o0 + wave * 16 + ((lane < 16) ? r : r + 8);
      int n = n0 + nt * 16 + (lane & 15);
      int h = o >> 5, dd = o & 31;
      qh[(((size_t)b * HEADS + h) * NTOK + n) * DHEAD + dd] = (_Float16)(acc[nt][r] * scale);
    }
  }
}

// ---------------------------------------------------------------------------
// KV: 4x4 stride-4 conv. With wkvP K-order (i,j,c), each tap (i,j) is a
// dense K=256 GEMM whose B-panel rows are contiguous xT rows.
// 32-stage async DOUBLE-BUFFERED pipeline (tap x half-panel): issue next
// stage's 8 async copies, wait asynccnt<=8, compute current out of LDS.
// k stored [b,h,m,d]; v stored TRANSPOSED [b,h,d,m].
// ---------------------------------------------------------------------------
__global__ __launch_bounds__(128) void kv_kernel(const _Float16* __restrict__ xT,
                                                 const _Float16* __restrict__ wkvP,
                                                 _Float16* __restrict__ kh,
                                                 _Float16* __restrict__ vTh) {
  const int b  = blockIdx.z;
  const int o0 = blockIdx.y * 64;   // over 512
  const int m0 = blockIdx.x * 64;   // over 256
  const int tid = threadIdx.x;
  const int wave = tid >> 5, lane = tid & 31;

  __shared__ __align__(16) _Float16 Bs[2][64][KVP];  // double-buffered half panels

  v8f acc[4] = {};
  const int arow = o0 + wave * 16 + (lane & 15);
  const int koff = (lane < 16) ? 0 : 8;
  const int kb   = (lane < 16) ? 0 : 16;

  // stage s: tap ij = s>>1, c-half cp = (s&1)*128; 1024 chunks, 8/thread
  auto stage = [&](int s) {
    const int ij = s >> 1;
    const int cp = (s & 1) * 128;
    const int i4 = ij >> 2, j4 = ij & 3;
    for (int ch = tid; ch < 1024; ch += 128) {
      int mm = ch >> 4;
      int cc = (ch & 15) * 8;
      int m  = m0 + mm;
      int my = m >> 4, mx = m & 15;
      int ns = (my * 4 + i4) * 64 + (mx * 4 + j4);
      async_b128(&Bs[s & 1][mm][cc],
                 &xT[((size_t)b * NTOK + ns) * CDIM + cp + cc]);
    }
  };

  stage(0);
  for (int s = 0; s < 32; ++s) {
    if (s < 31) {
      stage(s + 1);       // prefetch next half-panel into the other buffer
      wait_async_le8();   // oldest 8 (current panel) complete, next 8 in flight
    } else {
      wait_async0();
    }
    __syncthreads();

    const int ij = s >> 1;
    const int cp = (s & 1) * 128;
    const _Float16 (*buf)[KVP] = Bs[s & 1];
    for (int kt = 0; kt < 4; ++kt) {
      const _Float16* wp = wkvP + (size_t)arow * 4096 + ij * 256 + cp + kt * 32;
      v16h a = ld16(wp + kb, wp + kb + 8);
#pragma unroll
      for (int nt = 0; nt < 4; ++nt) {
        const _Float16* bp = &buf[nt * 16 + (lane & 15)][kt * 32 + kb];
        v16h bm = ld16(bp, bp + 8);
        acc[nt] = WMMA_F32_F16(a, bm, acc[nt]);
      }
    }
    __syncthreads();   // all waves done reading before this buffer is reused
  }

#pragma unroll
  for (int nt = 0; nt < 4; ++nt) {
#pragma unroll
    for (int r = 0; r < 8; ++r) {
      int o = o0 + wave * 16 + ((lane < 16) ? r : r + 8);
      int m = m0 + nt * 16 + (lane & 15);
      float val = acc[nt][r];
      if (o < CDIM) {
        int h = o >> 5, dd = o & 31;
        kh[(((size_t)b * HEADS + h) * MTOK + m) * DHEAD + dd] = (_Float16)val;
      } else {
        int o2 = o - CDIM;
        int h = o2 >> 5, dd = o2 & 31;
        vTh[(((size_t)b * HEADS + h) * DHEAD + dd) * MTOK + m] = (_Float16)val;
      }
    }
  }
}

// ---------------------------------------------------------------------------
// Fused attention: per (b,h), 16 q-rows per wave, full 256-wide sim in LDS,
// exact softmax, p*v via WMMA. 2 waves per WG. Padded LDS rows (4-bank
// spread) for sim reads and p A-operand reads.
// Output stored TRANSPOSED: aoT[b][n][c] (contiguous c across lanes).
// ---------------------------------------------------------------------------
__global__ __launch_bounds__(64) void attn_kernel(const _Float16* __restrict__ qh,
                                                  const _Float16* __restrict__ kh,
                                                  const _Float16* __restrict__ vTh,
                                                  _Float16* __restrict__ aoT) {
  const int b = blockIdx.z, h = blockIdx.y, nb = blockIdx.x;
  const int tid = threadIdx.x;
  const int wave = tid >> 5, lane = tid & 31;
  const int n0 = nb * 32 + wave * 16;
  const size_t bh = (size_t)b * HEADS + h;

  __shared__ __align__(16) float    simS[2][16][SIMP];
  __shared__ __align__(16) _Float16 pS[2][16][PSP];
  __shared__ float rowsum[2][16];

  const int koff = (lane < 16) ? 0 : 8;
  const int kb   = (lane < 16) ? 0 : 16;

  // q A-operand (K = d = 32 exactly: one WMMA per 16x16 sim tile)
  const _Float16* qp = qh + (bh * NTOK + (n0 + (lane & 15))) * DHEAD;
  v16h aq = ld16(qp + kb, qp + kb + 8);

  // sim = (q*scale) k^T  -> [16 x 256] in LDS
#pragma unroll 1
  for (int t = 0; t < 16; ++t) {
    const _Float16* kp = kh + (bh * MTOK + t * 16 + (lane & 15)) * DHEAD + kb;
    v16h bk = ld16(kp, kp + 8);
    v8f s = {};
    s = WMMA_F32_F16(aq, bk, s);
#pragma unroll
    for (int r = 0; r < 8; ++r) {
      int row = (lane < 16) ? r : r + 8;
      simS[wave][row][t * 16 + (lane & 15)] = s[r];
    }
  }
  __syncthreads();

  // exact softmax: 2 lanes per row, 128 cols each; store unnormalized exp,
  // fold 1/rowsum into the output store.
  {
    int row = lane >> 1;
    int cb = (lane & 1) * 128;
    float mx = -3.4e38f;
    for (int c = 0; c < 128; c += 4) {
      float4 sv = *(const float4*)&simS[wave][row][cb + c];
      mx = fmaxf(mx, fmaxf(fmaxf(sv.x, sv.y), fmaxf(sv.z, sv.w)));
    }
    mx = fmaxf(mx, __shfl_xor(mx, 1));
    float sum = 0.f;
    for (int c = 0; c < 128; c += 4) {
      float4 sv = *(const float4*)&simS[wave][row][cb + c];
      float e0 = __expf(sv.x - mx);
      float e1 = __expf(sv.y - mx);
      float e2 = __expf(sv.z - mx);
      float e3 = __expf(sv.w - mx);
      sum += (e0 + e1) + (e2 + e3);
      v4h pv = {(_Float16)e0, (_Float16)e1, (_Float16)e2, (_Float16)e3};
      *(v4h*)&pS[wave][row][cb + c] = pv;
    }
    sum += __shfl_xor(sum, 1);
    if ((lane & 1) == 0) rowsum[wave][row] = sum;
  }
  __syncthreads();

  // out = p v : [16 x 32], K = 256 in chunks of 32
#pragma unroll
  for (int dt = 0; dt < 2; ++dt) {
    v8f acc = {};
#pragma unroll 1
    for (int kt = 0; kt < 8; ++kt) {
      const _Float16* pp = &pS[wave][lane & 15][kt * 32];
      v16h ap = ld16(pp + kb, pp + kb + 8);
      const _Float16* vp = vTh + (bh * DHEAD + dt * 16 + (lane & 15)) * MTOK + kt * 32 + kb;
      v16h bv = ld16(vp, vp + 8);
      acc = WMMA_F32_F16(ap, bv, acc);
    }
#pragma unroll
    for (int r = 0; r < 8; ++r) {
      int rr = (lane < 16) ? r : r + 8;
      float val = acc[r] / rowsum[wave][rr];
      int cch = h * DHEAD + dt * 16 + (lane & 15);
      aoT[((size_t)b * NTOK + (n0 + rr)) * CDIM + cch] = (_Float16)val;
    }
  }
}

// ---------------------------------------------------------------------------
// Output projection: out[b,o,n] = sum_c w_out[o,c] * aoT[n,c]  (fp32 out)
// Same async full-panel structure as qproj.
// ---------------------------------------------------------------------------
__global__ __launch_bounds__(128) void outproj_kernel(const _Float16* __restrict__ aoT,
                                                      const _Float16* __restrict__ wouth,
                                                      float* __restrict__ out) {
  const int b  = blockIdx.z;
  const int o0 = blockIdx.y * 64;
  const int n0 = blockIdx.x * 64;
  const int tid = threadIdx.x;
  const int wave = tid >> 5, lane = tid & 31;

  __shared__ __align__(16) _Float16 Xs[64][LDSC];   // [n][c]

  for (int ch = tid; ch < 2048; ch += 128) {
    int row = ch >> 5;
    int cc  = (ch & 31) * 8;
    async_b128(&Xs[row][cc], &aoT[((size_t)b * NTOK + n0 + row) * CDIM + cc]);
  }
  wait_async0();
  __syncthreads();

  v8f acc[4] = {};
  const int arow = o0 + wave * 16 + (lane & 15);
  const int koff = (lane < 16) ? 0 : 8;
  const int kb   = (lane < 16) ? 0 : 16;

  for (int kc = 0; kc < CDIM; kc += 32) {
    const _Float16* wp = wouth + (size_t)arow * CDIM + kc;
    v16h a = ld16(wp + kb, wp + kb + 8);
#pragma unroll
    for (int nt = 0; nt < 4; ++nt) {
      const _Float16* bp = &Xs[nt * 16 + (lane & 15)][kc + kb];
      v16h bm = ld16(bp, bp + 8);
      acc[nt] = WMMA_F32_F16(a, bm, acc[nt]);
    }
  }

#pragma unroll
  for (int nt = 0; nt < 4; ++nt) {
#pragma unroll
    for (int r = 0; r < 8; ++r) {
      int o = o0 + wave * 16 + ((lane < 16) ? r : r + 8);
      int n = n0 + nt * 16 + (lane & 15);
      out[((size_t)b * CDIM + o) * NTOK + n] = acc[nt][r];
    }
  }
}

// ---------------------------------------------------------------------------
extern "C" void kernel_launch(void* const* d_in, const int* in_sizes, int n_in,
                              void* d_out, int out_size, void* d_ws, size_t ws_size,
                              hipStream_t stream) {
  const float* x     = (const float*)d_in[0];
  const float* w_q   = (const float*)d_in[1];
  const float* w_kv  = (const float*)d_in[2];
  const float* w_out = (const float*)d_in[3];
  float* out = (float*)d_out;

  char* p = (char*)d_ws;
  auto carve = [&](size_t elems) {
    _Float16* r = (_Float16*)p;
    p += (elems * sizeof(_Float16) + 255) & ~(size_t)255;
    return r;
  };
  _Float16* xT    = carve((size_t)BATCH * NTOK * CDIM);        // x^T fp16 [b][n][c]
  _Float16* wqh   = carve((size_t)CDIM * CDIM);
  _Float16* wkvP  = carve((size_t)2 * CDIM * CDIM * 16);       // [512][(i,j,c)]
  _Float16* wouth = carve((size_t)CDIM * CDIM);
  _Float16* qh    = carve((size_t)BATCH * HEADS * NTOK * DHEAD);
  _Float16* kh    = carve((size_t)BATCH * HEADS * MTOK * DHEAD);
  _Float16* vTh   = carve((size_t)BATCH * HEADS * DHEAD * MTOK);
  _Float16* aoT   = carve((size_t)BATCH * NTOK * CDIM);        // attn out [b][n][c]
  (void)ws_size; (void)in_sizes; (void)n_in; (void)out_size;

  transpose_cvt<<<dim3(NTOK / 32, CDIM / 32, BATCH), 256, 0, stream>>>(x, xT);
  cvt_f32_f16  <<<64, 256, 0, stream>>>(w_q, wqh, CDIM * CDIM);
  permute_wkv  <<<(2 * CDIM * CDIM * 16) / 256, 256, 0, stream>>>(w_kv, wkvP);
  cvt_f32_f16  <<<64, 256, 0, stream>>>(w_out, wouth, CDIM * CDIM);

  qproj_kernel  <<<dim3(NTOK / 64, CDIM / 64, BATCH), 128, 0, stream>>>(xT, wqh, qh);
  kv_kernel     <<<dim3(MTOK / 64, (2 * CDIM) / 64, BATCH), 128, 0, stream>>>(xT, wkvP, kh, vTh);
  attn_kernel   <<<dim3(NTOK / 32, HEADS, BATCH), 64, 0, stream>>>(qh, kh, vTh, aoT);
  outproj_kernel<<<dim3(NTOK / 64, CDIM / 64, BATCH), 128, 0, stream>>>(aoT, wouth, out);
}
